// LSTM_34943853920920
// MI455X (gfx1250) — compile-verified, matrix operand order checked by reading, not run on hardware
//
#include <hip/hip_runtime.h>

#define T_LEN 3000
#define H_DIM 32
#define B_SZ  1024
#define CT    60   // 3000 = 50 * 60, no remainder

typedef __attribute__((ext_vector_type(16))) _Float16 v16h;
typedef __attribute__((ext_vector_type(8)))  _Float16 v8h;
typedef __attribute__((ext_vector_type(8)))  float    v8f;

__device__ __forceinline__ float fast_sigmoid(float x) {
    return __builtin_amdgcn_rcpf(1.0f + __expf(-x));
}
__device__ __forceinline__ float fast_tanh(float x) {
    // tanh(x) = 1 - 2/(e^{2x}+1)
    return 1.0f - 2.0f * __builtin_amdgcn_rcpf(1.0f + __expf(2.0f * x));
}

// Build a 16x16x32 f16 WMMA A/B fragment from a row-major [16][32] f16 LDS tile.
// Layout (ISA 7.12.2, 16-bit A 16x32): lane m (0-15) holds row m, K = {0..7, 16..23};
// lane m+16 holds row m, K = {8..15, 24..31}. B uses the same pattern with
// "row" = output column (i.e. we stage B as rows of W_hh = columns of W_hh^T).
__device__ __forceinline__ v16h load_frag_rowmajor(const _Float16* base, int ln, int hi) {
    const _Float16* p = base + ln * 32 + hi * 8;
    v8h lo = *(const v8h*)(p);        // K = 0..7   (or 8..15 for upper half)
    v8h hh = *(const v8h*)(p + 16);   // K = 16..23 (or 24..31)
    return __builtin_shufflevector(lo, hh, 0,1,2,3,4,5,6,7,8,9,10,11,12,13,14,15);
}

__global__ __launch_bounds__(32) void lstm_wmma_kernel(
    const float* __restrict__ x,    const float* __restrict__ W_ih,
    const float* __restrict__ W_hh, const float* __restrict__ b_ih,
    const float* __restrict__ b_hh, const float* __restrict__ W3,
    const float* __restrict__ b3,   const float* __restrict__ W4,
    const float* __restrict__ b4,   float* __restrict__ out)
{
    __shared__ __align__(16) _Float16 stage[16 * 32];  // B-fragment staging
    __shared__ __align__(16) _Float16 hbuf [16 * 32];  // h (f16) D->A transpose buffer
    __shared__ __align__(16) float    xbuf [16 * CT];  // x chunk, row-major [16][CT]
    __shared__ __align__(16) float    hlast[16 * 32];  // f32 h at t = T-1

    const int lane = threadIdx.x;     // wave32: lane id
    const int ln   = lane & 15;
    const int hi   = lane >> 4;       // lane half
    const int m0   = blockIdx.x * 16; // batch tile base row

    // ---- Load W_hh^T once into 8 resident B fragments (one per 16 gate cols) ----
    v16h b_frag[8];
    for (int g = 0; g < 8; ++g) {
        for (int idx = lane; idx < 512; idx += 32) {
            int r = idx >> 5, k = idx & 31;
            // B[k][n] = W_hh^T[k][16g+n] = W_hh[16g+n][k]; stage "row n" = W_hh row.
            stage[idx] = (_Float16)W_hh[(16 * g + r) * H_DIM + k];
        }
        __syncthreads();
        b_frag[g] = load_frag_rowmajor(stage, ln, hi);
        __syncthreads();
    }

    // Per-lane column constants: combined bias and W_ih (input dim == 1).
    float bias[8], wih[8];
    #pragma unroll
    for (int g = 0; g < 8; ++g) {
        int col = 16 * g + ln;
        bias[g] = b_ih[col] + b_hh[col];
        wih[g]  = W_ih[col];
    }

    v16h a_h    = {};   // h0 = 0 in A layout
    v8f  cst[2] = {};   // c state, accumulator layout (2 tiles of 16 cols)

    for (int t0 = 0; t0 < T_LEN; t0 += CT) {
        __syncthreads();
        // Stage x[m0..m0+15, t0..t0+CT) into LDS (coalesced along T).
        for (int idx = lane; idx < 16 * CT; idx += 32) {
            int r = idx / CT;
            int c = idx - r * CT;
            xbuf[idx] = x[(size_t)(m0 + r) * T_LEN + t0 + c];
        }
        __syncthreads();

        for (int tc = 0; tc < CT; ++tc) {
            // x value per accumulator row (row = r + 8*hi); LDS broadcast reads.
            float xv[8];
            #pragma unroll
            for (int r = 0; r < 8; ++r)
                xv[r] = xbuf[(r + 8 * hi) * CT + tc];

            // z = [x*W_ih + b] + h @ W_hh^T : 8 WMMAs, x-proj folded into C.
            v8f z[8];
            #pragma unroll
            for (int g = 0; g < 8; ++g) {
                v8f c;
                #pragma unroll
                for (int r = 0; r < 8; ++r)
                    c[r] = __builtin_fmaf(xv[r], wih[g], bias[g]);
                z[g] = __builtin_amdgcn_wmma_f32_16x16x32_f16(
                    false, a_h, false, b_frag[g], (short)0, c, false, false);
            }

            // Gate order i,f,g,o -> z tiles {0,1},{2,3},{4,5},{6,7}. h has 2 col tiles.
            #pragma unroll
            for (int j = 0; j < 2; ++j) {
                #pragma unroll
                for (int r = 0; r < 8; ++r) {
                    float iv = fast_sigmoid(z[0 + j][r]);
                    float fv = fast_sigmoid(z[2 + j][r]);
                    float gv = fast_tanh  (z[4 + j][r]);
                    float ov = fast_sigmoid(z[6 + j][r]);
                    float cn = __builtin_fmaf(fv, cst[j][r], iv * gv);
                    cst[j][r] = cn;
                    float hn = ov * fast_tanh(cn);
                    // D layout: VGPR r = row (r + 8*hi), lane ln = col within tile j.
                    hbuf[(r + 8 * hi) * 32 + 16 * j + ln] = (_Float16)hn;
                    if (t0 + tc == T_LEN - 1)        // uniform scalar branch
                        hlast[(r + 8 * hi) * 32 + 16 * j + ln] = hn;
                }
            }
            __syncthreads();   // single-wave WG: s_wait_dscnt + s_nop barrier
            a_h = load_frag_rowmajor(hbuf, ln, hi);
        }
    }

    __syncthreads();
    // ---- Tiny head: relu(h @ W3^T + b3) @ W4^T + b4, sigmoid. One row per lane. ----
    if (lane < 16) {
        float s = b4[0];
        for (int j = 0; j < 32; ++j) {
            float a = b3[j];
            #pragma unroll
            for (int k = 0; k < 32; ++k)
                a = __builtin_fmaf(hlast[lane * 32 + k], W3[j * 32 + k], a);
            a = fmaxf(a, 0.0f);
            s = __builtin_fmaf(a, W4[j], s);
        }
        out[m0 + lane] = fast_sigmoid(s);
    }
}

extern "C" void kernel_launch(void* const* d_in, const int* in_sizes, int n_in,
                              void* d_out, int out_size, void* d_ws, size_t ws_size,
                              hipStream_t stream) {
    const float* x    = (const float*)d_in[0];
    const float* W_ih = (const float*)d_in[1];
    const float* W_hh = (const float*)d_in[2];
    const float* b_ih = (const float*)d_in[3];
    const float* b_hh = (const float*)d_in[4];
    const float* W3   = (const float*)d_in[5];
    const float* b3   = (const float*)d_in[6];
    const float* W4   = (const float*)d_in[7];
    const float* b4   = (const float*)d_in[8];
    float* outp = (float*)d_out;
    (void)in_sizes; (void)n_in; (void)out_size; (void)d_ws; (void)ws_size;

    lstm_wmma_kernel<<<B_SZ / 16, 32, 0, stream>>>(
        x, W_ih, W_hh, b_ih, b_hh, W3, b3, W4, b4, outp);
}
